// DKM_3874060501119
// MI455X (gfx1250) — matrix operand order repaired and checked by reading, not compiled, not run on hardware
//
#include <hip/hip_runtime.h>
#include <hip/hip_bf16.h>
#include <math.h>

// ---------------------------------------------------------------------------
// DKM (soft k-means) for MI455X / gfx1250, wave32, bf16 WMMA (16x16x32, f32 acc)
// + Tensor Data Mover (TDM) double-buffered LDS staging for the attn^T @ W GEMM
// ---------------------------------------------------------------------------

#define NS   65536   // samples
#define DD   256     // feature dim
#define KC   512     // clusters
#define ITERS 10
#define EPS  1e-4f

typedef __bf16  bf16_t;
typedef __bf16  v16bf __attribute__((ext_vector_type(16)));
typedef __bf16  v8bf  __attribute__((ext_vector_type(8)));
typedef float   v8f   __attribute__((ext_vector_type(8)));
typedef unsigned int v4u __attribute__((ext_vector_type(4)));
typedef unsigned int v8u __attribute__((ext_vector_type(8)));

// ---- workspace layout (bytes) ----
constexpr size_t OFF_WBF   = 0;                                  // N*D bf16
constexpr size_t OFF_WT    = OFF_WBF   + (size_t)NS * DD * 2;    // D*N bf16 (W^T)
constexpr size_t OFF_ATTN  = OFF_WT    + (size_t)NS * DD * 2;    // N*K bf16
constexpr size_t OFF_ATTNT = OFF_ATTN  + (size_t)NS * KC * 2;    // K*N bf16 (attn^T)
constexpr size_t OFF_CCUR  = OFF_ATTNT + (size_t)NS * KC * 2;    // K*D f32
constexpr size_t OFF_CNXT  = OFF_CCUR  + (size_t)KC * DD * 4;    // K*D f32
constexpr size_t OFF_CBF   = OFF_CNXT  + (size_t)KC * DD * 4;    // K*D bf16
constexpr size_t OFF_CT    = OFF_CBF   + (size_t)KC * DD * 2;    // D*K bf16 (C^T)
constexpr size_t OFF_WSUM  = OFF_CT    + (size_t)KC * DD * 2;    // K*D f32
constexpr size_t OFF_CNT   = OFF_WSUM  + (size_t)KC * DD * 4;    // K f32
constexpr size_t OFF_WNORM = OFF_CNT   + (size_t)KC * 4;         // N f32
constexpr size_t OFF_CNORM = OFF_WNORM + (size_t)NS * 4;         // K f32
constexpr size_t OFF_FLAGS = OFF_CNORM + (size_t)KC * 4;         // norm2,done,do_update

// ---------------------------------------------------------------------------
// WMMA fragment helpers per CDNA5 ISA 7.12.2 per-lane layouts (wave32).
// A 16x32 bf16: lane m=lane&15; k in {kb..kb+7} U {16+kb..23+kb}, kb = lane<16?0:8
// B 32x16 bf16: lane n=lane&15; k in {kb..kb+7} U {16+kb..23+kb}, kb = lane<16?0:8
//   (matrix stored "column-major": memory layout [n][k], contiguous along k)
// ---------------------------------------------------------------------------
static __device__ __forceinline__ v16bf load_frag_a(const bf16_t* __restrict__ base,
                                                    long lda, int m0, long k0, int lane) {
    int m  = m0 + (lane & 15);
    int kb = (lane < 16) ? 0 : 8;
    const bf16_t* p = base + (long)m * lda + k0;
    v8bf lo = *(const v8bf*)(p + kb);
    v8bf hi = *(const v8bf*)(p + 16 + kb);
    return __builtin_shufflevector(lo, hi, 0,1,2,3,4,5,6,7,8,9,10,11,12,13,14,15);
}

static __device__ __forceinline__ v16bf load_frag_b(const bf16_t* __restrict__ base,
                                                    long ldb, int n0, long k0, int lane) {
    int n  = n0 + (lane & 15);
    int kb = (lane < 16) ? 0 : 8;
    const bf16_t* p = base + (long)n * ldb + k0 + kb;
    v8bf lo = *(const v8bf*)(p);
    v8bf hi = *(const v8bf*)(p + 16);
    return __builtin_shufflevector(lo, hi, 0,1,2,3,4,5,6,7,8,9,10,11,12,13,14,15);
}

static __device__ __forceinline__ v8f wmma_bf16(v16bf a, v16bf b, v8f c) {
    // (neg_a, A, neg_b, B, c_mod, C, reuse_a, reuse_b)
    return __builtin_amdgcn_wmma_f32_16x16x32_bf16(false, a, false, b, (short)0, c,
                                                   false, false);
}

// ---------------------------------------------------------------------------
// TDM: issue a 2D bf16 tile load Global -> LDS (one wave issues; EXEC ignored).
// D# bit packing per CDNA5 ISA (group0) / §8.4 (group1).
//   group0: [1:0]=count=1, [63:32]=lds_addr, [120:64]=global_addr, [127:126]=type=2
//   group1: [17:16]=data_size=1 (2B), tensor_dim0/1, tile_dim0/1,
//           tensor_dim0_stride (48b, data_size units)
// LDS byte offset = low 32 bits of the generic pointer (LDS aperture, ISA 10.2).
// ---------------------------------------------------------------------------
static __device__ __forceinline__ void tdm_load_tile_bf16(void* lds_dst,
                                                          const void* gsrc,
                                                          unsigned tile_w,    // elems (contig)
                                                          unsigned tile_h,    // rows
                                                          unsigned long long row_stride) { // elems
    unsigned long long ga = (unsigned long long)gsrc;
    unsigned lds_off = (unsigned)(unsigned long long)lds_dst;   // addr[31:0] = LDS offset
    v4u g0;
    g0[0] = 1u;                                                 // count=1, user descriptor
    g0[1] = lds_off;
    g0[2] = (unsigned)ga;
    g0[3] = (unsigned)((ga >> 32) & 0x01FFFFFFu) | (2u << 30);  // global_addr[56:32] | type=2
    v8u g1;
    g1[0] = (1u << 16);                                         // wg_mask=0, data_size=1 (2B)
    g1[1] = (tile_w & 0xFFFFu) << 16;                           // tensor_dim0[15:0]
    g1[2] = ((tile_w >> 16) & 0xFFFFu) | ((tile_h & 0xFFFFu) << 16); // tdim0 hi | tensor_dim1 lo
    g1[3] = ((tile_h >> 16) & 0xFFFFu) | ((tile_w & 0xFFFFu) << 16); // tdim1 hi | tile_dim0
    g1[4] = (tile_h & 0xFFFFu);                                 // tile_dim1 (tile_dim2=0)
    g1[5] = (unsigned)(row_stride & 0xFFFFFFFFull);             // tensor_dim0_stride[31:0]
    g1[6] = (unsigned)((row_stride >> 32) & 0xFFFFull);         // stride[47:32] | dim1_stride lo=0
    g1[7] = 0u;
    asm volatile("tensor_load_to_lds %0, %1" :: "s"(g0), "s"(g1) : "memory");
}

// ---------------------------------------------------------------------------
// Preprocess weights: bf16 row-major, bf16 transposed, per-row squared norm.
// ---------------------------------------------------------------------------
__global__ __launch_bounds__(256) void k_prep_weights(const float* __restrict__ w,
                                                      bf16_t* __restrict__ wbf,
                                                      bf16_t* __restrict__ wt,
                                                      float* __restrict__ wnorm) {
    __shared__ float red[256];
    long n = blockIdx.x;
    int  t = threadIdx.x;
    float v = w[n * DD + t];
    bf16_t b = (bf16_t)v;
    wbf[n * DD + t] = b;
    wt[(long)t * NS + n] = b;
    red[t] = v * v;
    __syncthreads();
    for (int s = 128; s > 0; s >>= 1) {
        if (t < s) red[t] += red[t + s];
        __syncthreads();
    }
    if (t == 0) wnorm[n] = red[0];
}

__global__ __launch_bounds__(256) void k_prep_clusters(const float* __restrict__ cur,
                                                       bf16_t* __restrict__ cbf,
                                                       float* __restrict__ cnorm,
                                                       const int* __restrict__ done) {
    if (*done) return;
    __shared__ float red[256];
    long k = blockIdx.x;
    int  t = threadIdx.x;
    float v = cur[k * DD + t];
    cbf[k * DD + t] = (bf16_t)v;
    red[t] = v * v;
    __syncthreads();
    for (int s = 128; s > 0; s >>= 1) {
        if (t < s) red[t] += red[t + s];
        __syncthreads();
    }
    if (t == 0) cnorm[k] = red[0];
}

// ---------------------------------------------------------------------------
// Fused: logits = -sqrt(max(|w|^2 - 2 W@C^T + |c|^2, 1e-12)); softmax over K.
// 64 rows x 512 clusters per block, 16 waves (4Mx4N), ~134 KB dynamic LDS.
// ---------------------------------------------------------------------------
#define LDL (KC + 4)
__global__ __launch_bounds__(512) void k_dist_softmax(const bf16_t* __restrict__ wbf,
                                                      const bf16_t* __restrict__ cbf,
                                                      const float* __restrict__ wnorm,
                                                      const float* __restrict__ cnorm,
                                                      float* __restrict__ attn_f32,
                                                      bf16_t* __restrict__ attn_bf,
                                                      bf16_t* __restrict__ attn_T,
                                                      const int* __restrict__ done) {
    if (*done) return;
    extern __shared__ char smem_raw[];
    float* logits = (float*)smem_raw;             // [64][LDL]
    float* red    = logits + 64 * LDL;            // [64][8]
    float* rowval = red + 64 * 8;                 // [64]
    float* s_wn   = rowval + 64;                  // [64]
    float* s_cn   = s_wn + 64;                    // [512]

    const int t    = threadIdx.x;
    const int lane = t & 31;
    const int wave = t >> 5;
    const int mw   = wave >> 2;   // 0..3
    const int nw   = wave & 3;    // 0..3
    const int m0   = blockIdx.x * 64;

    if (t < 64) s_wn[t] = wnorm[m0 + t];
    s_cn[t] = cnorm[t];
    __syncthreads();

    v8f acc[8];
#pragma unroll
    for (int j = 0; j < 8; ++j) acc[j] = (v8f){0.f,0.f,0.f,0.f,0.f,0.f,0.f,0.f};

    const int mrow  = m0 + mw * 16;
    const int nbase = nw * 128;
    for (int ks = 0; ks < DD; ks += 32) {
        v16bf av = load_frag_a(wbf, DD, mrow, ks, lane);
#pragma unroll
        for (int j = 0; j < 8; ++j) {
            v16bf bv = load_frag_b(cbf, DD, nbase + j * 16, ks, lane);
            acc[j] = wmma_bf16(av, bv, acc[j]);
        }
    }

    // C layout: VGPR r, lanes 0-15: M=r, lanes 16-31: M=8+r; N = lane&15
    const int rbase = mw * 16 + (lane >> 4) * 8;
    const int cl    = lane & 15;
#pragma unroll
    for (int j = 0; j < 8; ++j) {
        int   col = nbase + j * 16 + cl;
        float cn  = s_cn[col];
#pragma unroll
        for (int r = 0; r < 8; ++r) {
            int   row = rbase + r;
            float d2  = s_wn[row] + cn - 2.0f * acc[j][r];
            d2 = fmaxf(d2, 1e-12f);
            logits[row * LDL + col] = -sqrtf(d2);   // TEMPERATURE == 1
        }
    }
    __syncthreads();

    // softmax over 512 cols: thread -> (row = t>>3, 64-col chunk = t&7)
    const int row = t >> 3;
    const int ch  = t & 7;
    const int c0  = ch * 64;
    float mx = -1e30f;
    for (int c = 0; c < 64; ++c) mx = fmaxf(mx, logits[row * LDL + c0 + c]);
    red[row * 8 + ch] = mx;
    __syncthreads();
    if (ch == 0) {
        float mm = red[row * 8];
        for (int i = 1; i < 8; ++i) mm = fmaxf(mm, red[row * 8 + i]);
        rowval[row] = mm;
    }
    __syncthreads();
    const float rmax = rowval[row];
    float s = 0.f;
    for (int c = 0; c < 64; ++c) {
        float e = __expf(logits[row * LDL + c0 + c] - rmax);
        logits[row * LDL + c0 + c] = e;
        s += e;
    }
    red[row * 8 + ch] = s;
    __syncthreads();
    if (ch == 0) {
        float ss = red[row * 8];
        for (int i = 1; i < 8; ++i) ss += red[row * 8 + i];
        rowval[row] = ss;
    }
    __syncthreads();

    // row-major outputs (f32 to d_out, bf16 for final GEMM)
    {
        const float inv  = 1.0f / rowval[row];
        const long  base = (long)(m0 + row) * KC + c0;
        for (int c = 0; c < 64; ++c) {
            float a = logits[row * LDL + c0 + c] * inv;
            attn_f32[base + c] = a;
            attn_bf[base + c]  = (bf16_t)a;
        }
    }
    // transposed bf16 (for the attn^T @ W GEMM): thread t owns cluster t
    {
        const long base = (long)t * NS + m0;
        for (int r = 0; r < 64; ++r) {
            float a = logits[r * LDL + t] / rowval[r];
            attn_T[base + r] = (bf16_t)a;
        }
    }
}

// ---------------------------------------------------------------------------
// weighted_sum += attn^T @ W (K x D), split over N with f32 global atomics.
// TDM double-buffered LDS staging: wave 0 issues two 64x64 bf16 tile DMAs per
// step (TENSORcnt-tracked), all 16 waves consume fragments from LDS.
// grid (K/64, D/64, 32), 512 threads, wave grid 4x4, each wave one 16x16 tile.
// ---------------------------------------------------------------------------
__global__ __launch_bounds__(512) void k_accum(const bf16_t* __restrict__ attn_T,
                                               const bf16_t* __restrict__ wt,
                                               float* __restrict__ wsum,
                                               float* __restrict__ counts,
                                               const int* __restrict__ done) {
    if (*done) return;
    __shared__ bf16_t As[2][64 * 64];   // [buf][row=cluster][k=sample]
    __shared__ bf16_t Bs[2][64 * 64];   // [buf][row=dim][k=sample]

    const int t     = threadIdx.x;
    const int lane  = t & 31;
    const int wave  = t >> 5;
    const int mw    = wave >> 2;
    const int nw    = wave & 3;
    const int krow0 = blockIdx.x * 64;            // cluster block
    const int dcol0 = blockIdx.y * 64;            // dim block
    const long n0   = (long)blockIdx.z * 2048;    // sample chunk
    const int STEPS = 2048 / 64;                  // 32 tiles of 64 samples

    if (wave == 0) {
        tdm_load_tile_bf16(&As[0][0], attn_T + (long)krow0 * NS + n0, 64, 64, NS);
        tdm_load_tile_bf16(&Bs[0][0], wt + (long)dcol0 * NS + n0, 64, 64, NS);
    }

    v8f   acc   = (v8f){0.f,0.f,0.f,0.f,0.f,0.f,0.f,0.f};
    float cpart = 0.f;
    for (int s = 0; s < STEPS; ++s) {
        const int cur = s & 1;
        if (wave == 0) {
            if (s + 1 < STEPS) {
                long nn = n0 + (long)(s + 1) * 64;
                tdm_load_tile_bf16(&As[cur ^ 1][0], attn_T + (long)krow0 * NS + nn, 64, 64, NS);
                tdm_load_tile_bf16(&Bs[cur ^ 1][0], wt + (long)dcol0 * NS + nn, 64, 64, NS);
                __builtin_amdgcn_s_wait_tensorcnt((short)2);  // current buffer ready
            } else {
                __builtin_amdgcn_s_wait_tensorcnt((short)0);
            }
        }
        __syncthreads();                          // LDS tile visible to all waves
#pragma unroll
        for (int ko = 0; ko < 64; ko += 32) {
            v16bf av = load_frag_a(&As[cur][0], 64, mw * 16, ko, lane);
            v16bf bv = load_frag_b(&Bs[cur][0], 64, nw * 16, ko, lane);
            acc = wmma_bf16(av, bv, acc);
            if (nw == 0) {                        // fold counts = attn column sums
#pragma unroll
                for (int i = 0; i < 16; ++i) cpart += (float)av[i];
            }
        }
        __syncthreads();                          // done with buffer before DMA reuse
    }

    const int mrow = krow0 + mw * 16;
    const int ncol = dcol0 + nw * 16;
    const int rb   = (lane >> 4) * 8;
    const int cl   = lane & 15;
#pragma unroll
    for (int r = 0; r < 8; ++r)
        atomicAdd(&wsum[(long)(mrow + rb + r) * DD + ncol + cl], acc[r]);
    if (nw == 0 && blockIdx.y == 0)
        atomicAdd(&counts[mrow + (lane & 15)], cpart);  // lanes L, L+16 cover disjoint k
}

// new_clusters = wsum / counts; accumulate ||cur - new||^2
__global__ __launch_bounds__(256) void k_update(const float* __restrict__ wsum,
                                                const float* __restrict__ counts,
                                                const float* __restrict__ cur,
                                                float* __restrict__ nxt,
                                                float* __restrict__ norm2,
                                                const int* __restrict__ done) {
    if (*done) return;
    __shared__ float red[256];
    const int  t = threadIdx.x;
    const long i = (long)blockIdx.x * 256 + t;
    const int  k = (int)(i >> 8);                 // DD == 256
    float nc = wsum[i] / counts[k];
    nxt[i] = nc;
    float d = cur[i] - nc;
    red[t] = d * d;
    __syncthreads();
    for (int s = 128; s > 0; s >>= 1) {
        if (t < s) red[t] += red[t + s];
        __syncthreads();
    }
    if (t == 0) atomicAdd(norm2, red[0]);
}

// scan semantics: small => freeze clusters at OLD value, done |= small
__global__ void k_check(const float* __restrict__ norm2, int* __restrict__ done,
                        int* __restrict__ do_update) {
    int   d  = *done;
    float n2 = *norm2;
    int small = (sqrtf(n2) <= EPS) ? 1 : 0;
    *do_update = (!d && !small) ? 1 : 0;
    if (!d && small) *done = 1;
}

__global__ __launch_bounds__(256) void k_copy_clusters(float* __restrict__ cur,
                                                       const float* __restrict__ nxt,
                                                       const int* __restrict__ do_update) {
    if (!*do_update) return;
    long i = (long)blockIdx.x * 256 + threadIdx.x;
    cur[i] = nxt[i];
}

// final cluster bf16 transpose [D][K] + copy clusters into d_out
__global__ __launch_bounds__(256) void k_prep_ct(const float* __restrict__ cur,
                                                 bf16_t* __restrict__ ct,
                                                 float* __restrict__ cl_out) {
    long i = (long)blockIdx.x * 256 + threadIdx.x;   // K*D elements
    int  k = (int)(i >> 8);
    int  d = (int)(i & 255);
    float v = cur[i];
    ct[(long)d * KC + k] = (bf16_t)v;
    cl_out[i] = v;
}

// compressed = attn (N x K) @ clusters (K x D); grid (N/64, D/64)
__global__ __launch_bounds__(512) void k_compress(const bf16_t* __restrict__ attn_bf,
                                                  const bf16_t* __restrict__ ct,
                                                  float* __restrict__ out) {
    const int t    = threadIdx.x;
    const int lane = t & 31;
    const int wave = t >> 5;
    const int mw   = wave >> 2;
    const int nw   = wave & 3;
    const int m0   = blockIdx.x * 64 + mw * 16;   // sample
    const int d0   = blockIdx.y * 64 + nw * 16;   // dim

    v8f acc = (v8f){0.f,0.f,0.f,0.f,0.f,0.f,0.f,0.f};
    for (int ks = 0; ks < KC; ks += 32) {
        __builtin_prefetch(attn_bf + (long)m0 * KC + ks + 128, 0, 1);
        v16bf av = load_frag_a(attn_bf, KC, m0, ks, lane);
        v16bf bv = load_frag_b(ct, KC, d0, ks, lane);
        acc = wmma_bf16(av, bv, acc);
    }
    const int rb = (lane >> 4) * 8;
    const int cl = lane & 15;
#pragma unroll
    for (int r = 0; r < 8; ++r)
        out[(long)(m0 + rb + r) * DD + d0 + cl] = acc[r];
}

// ---------------------------------------------------------------------------
extern "C" void kernel_launch(void* const* d_in, const int* in_sizes, int n_in,
                              void* d_out, int out_size, void* d_ws, size_t ws_size,
                              hipStream_t stream) {
    const float* w  = (const float*)d_in[0];
    char*        ws = (char*)d_ws;

    bf16_t* wbf    = (bf16_t*)(ws + OFF_WBF);
    bf16_t* wt     = (bf16_t*)(ws + OFF_WT);
    bf16_t* attnbf = (bf16_t*)(ws + OFF_ATTN);
    bf16_t* attnT  = (bf16_t*)(ws + OFF_ATTNT);
    float*  ccur   = (float*)(ws + OFF_CCUR);
    float*  cnxt   = (float*)(ws + OFF_CNXT);
    bf16_t* cbf    = (bf16_t*)(ws + OFF_CBF);
    bf16_t* ct     = (bf16_t*)(ws + OFF_CT);
    float*  wsum   = (float*)(ws + OFF_WSUM);
    float*  cnt    = (float*)(ws + OFF_CNT);
    float*  wnorm  = (float*)(ws + OFF_WNORM);
    float*  cnorm  = (float*)(ws + OFF_CNORM);
    float*  norm2  = (float*)(ws + OFF_FLAGS);
    int*    done   = (int*)(ws + OFF_FLAGS + 4);
    int*    do_upd = (int*)(ws + OFF_FLAGS + 8);

    float* out          = (float*)d_out;
    float* out_comp     = out;                                  // N*D
    float* out_clusters = out + (size_t)NS * DD;                // K*D
    float* out_attn     = out_clusters + (size_t)KC * DD;       // N*K

    const size_t SMEM1 = (size_t)(64 * LDL + 64 * 8 + 64 + 64 + 512) * sizeof(float);

    hipMemsetAsync(ws + OFF_FLAGS, 0, 12, stream);
    hipMemcpyAsync(ccur, w, (size_t)KC * DD * sizeof(float),
                   hipMemcpyDeviceToDevice, stream);
    k_prep_weights<<<NS, 256, 0, stream>>>(w, wbf, wt, wnorm);

    for (int it = 0; it < ITERS; ++it) {
        hipMemsetAsync(wsum, 0, (size_t)KC * DD * sizeof(float), stream);
        hipMemsetAsync(cnt, 0, (size_t)KC * sizeof(float), stream);
        hipMemsetAsync(norm2, 0, sizeof(float), stream);
        k_prep_clusters<<<KC, 256, 0, stream>>>(ccur, cbf, cnorm, done);
        k_dist_softmax<<<NS / 64, 512, SMEM1, stream>>>(wbf, cbf, wnorm, cnorm,
                                                        out_attn, attnbf, attnT, done);
        k_accum<<<dim3(KC / 64, DD / 64, 32), 512, 0, stream>>>(attnT, wt, wsum, cnt, done);
        k_update<<<KC * DD / 256, 256, 0, stream>>>(wsum, cnt, ccur, cnxt, norm2, done);
        k_check<<<1, 1, 0, stream>>>(norm2, done, do_upd);
        k_copy_clusters<<<KC * DD / 256, 256, 0, stream>>>(ccur, cnxt, do_upd);
    }

    k_prep_ct<<<KC * DD / 256, 256, 0, stream>>>(ccur, ct, out_clusters);
    k_compress<<<dim3(NS / 64, DD / 64), 512, 0, stream>>>(attnbf, ct, out_comp);
}
